// RobustPPRGoEmmbeddingDiffusions_59296318488774
// MI455X (gfx1250) — compile-verified, matrix operand order checked by reading, not run on hardware
//
#include <hip/hip_runtime.h>
#include <hip/hip_bf16.h>
#include <math.h>

// ---------- problem constants (from reference) ----------
#define NROWS 131072
#define NFEAT 512
#define HDIM  256
#define NCLS  128

typedef __bf16   v16bf __attribute__((ext_vector_type(16)));
typedef float    v8f   __attribute__((ext_vector_type(8)));
typedef unsigned v4u   __attribute__((ext_vector_type(4)));
typedef int      v8i   __attribute__((ext_vector_type(8)));
typedef int      v4i   __attribute__((ext_vector_type(4)));

#if defined(__has_builtin)
#if __has_builtin(__builtin_amdgcn_tensor_load_to_lds)
#define HAVE_TDM 1
#endif
#endif

// ---------- helpers ----------
__device__ __forceinline__ unsigned pack2bf(float a, float b) {
    union { __bf16 h[2]; unsigned u; } p;
    p.h[0] = (__bf16)a; p.h[1] = (__bf16)b;
    return p.u;
}
__device__ __forceinline__ unsigned ord_u32(float f) {
    unsigned u = __float_as_uint(f);
    return (u >> 31) ? ~u : (u | 0x80000000u);
}
__device__ __forceinline__ float ord_inv(unsigned u) {
    unsigned v = (u >> 31) ? (u ^ 0x80000000u) : ~u;
    return __uint_as_float(v);
}
__device__ __forceinline__ float blockReduceSum(float v, float* sm) {
    int t = threadIdx.x;
    sm[t] = v; __syncthreads();
    for (int s = 128; s > 0; s >>= 1) { if (t < s) sm[t] += sm[t + s]; __syncthreads(); }
    float r = sm[0]; __syncthreads();
    return r;
}
__device__ __forceinline__ float blockReduceMin(float v, float* sm) {
    int t = threadIdx.x;
    sm[t] = v; __syncthreads();
    for (int s = 128; s > 0; s >>= 1) { if (t < s) sm[t] = fminf(sm[t], sm[t + s]); __syncthreads(); }
    float r = sm[0]; __syncthreads();
    return r;
}

#ifdef HAVE_TDM
// Issue a 1-D TDM copy of `nwords` dwords from global -> LDS (wave-level op).
// D# per CDNA5 ISA 8.3/8.4: group0 {count=1, lds_addr, global_addr, type=2},
// group1 {data_size=4B, tensor_dim0=nwords, tensor_dim1=1, tile_dim0=nwords,
//         tile_dim1=1, tensor_dim0_stride=nwords}.
// This toolchain exposes the 6-arg builtin:
//   (uint32x4 g0, int32x8 g1, int32x4 g2, int32x4 g3, int32x8 extra, i32 cpol)
__device__ __forceinline__ void tdm_load_dwords(unsigned lds_off, const void* gptr,
                                                unsigned nwords) {
    unsigned long long ga = (unsigned long long)(uintptr_t)gptr;
    v4u g0;
    g0[0] = 1u;                                   // count=1, user mode
    g0[1] = lds_off;                              // LDS byte address
    g0[2] = (unsigned)(ga & 0xFFFFFFFFull);       // global_addr[31:0]
    g0[3] = (unsigned)((ga >> 32) & 0x1FFFFFFull) | (2u << 30); // ga[56:32] | type=2
    v8i g1;
    g1[0] = (int)(2u << 16);                      // workgroup_mask=0, data_size=4B
    g1[1] = (int)((nwords & 0xFFFFu) << 16);      // tensor_dim0[15:0]
    g1[2] = (int)((nwords >> 16) | (1u << 16));   // tensor_dim0[31:16] | tensor_dim1=1
    g1[3] = (int)((nwords & 0xFFFFu) << 16);      // tile_dim0 (16-bit)
    g1[4] = 1;                                    // tile_dim1=1, tile_dim2=0
    g1[5] = (int)nwords;                          // tensor_dim0_stride[31:0]
    g1[6] = 0;
    g1[7] = 0;
    v4i z4 = {0, 0, 0, 0};
    v8i z8 = {0, 0, 0, 0, 0, 0, 0, 0};
    __builtin_amdgcn_tensor_load_to_lds(g0, g1, z4, z4, z8, 0);
}
#endif

// ---------- kernel 0: init scalars ----------
__global__ void k_init(float* total_w, float* S, unsigned* dmin_bits) {
    if (threadIdx.x == 0 && blockIdx.x == 0) {
        *total_w = 0.0f;
        *S = 0.0f;
        *dmin_bits = 0x7F800000u; // +inf
    }
}

// ---------- kernel 1: sum of weights ----------
__global__ void k_wsum(const float* __restrict__ w, float* total_w) {
    __shared__ float sm[256];
    float acc = 0.0f;
    for (int i = blockIdx.x * 256 + threadIdx.x; i < NROWS; i += gridDim.x * 256)
        acc += w[i];
    float b = blockReduceSum(acc, sm);
    if (threadIdx.x == 0) atomicAdd(total_w, b);
}

// ---------- kernel 2: W -> fragment-ready packed-bf16-pair layout ----------
// Wf[chunk][tile][n][pr] = pack(W[chunk*32+2*pr][tile*16+n], W[chunk*32+2*pr+1][...])
// Each 16KB chunk is contiguous -> TDM-stageable; each lane's B fragment is two
// contiguous uint4 groups (pairRows {0..3,8..11} or {4..7,12..15}).
__global__ void k_cvtfrag(const float* __restrict__ W, unsigned* __restrict__ Wf, int K) {
    int o = blockIdx.x * 256 + threadIdx.x;
    int total = (K / 32) * 4096;
    if (o >= total) return;
    int pr = o & 15, n = (o >> 4) & 15, tile = (o >> 8) & 15, chunk = o >> 12;
    int k0 = chunk * 32 + 2 * pr;
    int c = tile * 16 + n;
    Wf[o] = pack2bf(W[(size_t)k0 * HDIM + c], W[(size_t)(k0 + 1) * HDIM + c]);
}

// ---------- kernel 3: fused E_T = transpose( relu(X@W1) @ W2 ) ----------
// 256 threads = 8 waves; 64 rows x 256 cols per block.
// wave (wr, wc): wr = wave>>1 rows [wr*16, +16), wc = wave&1 col half (8 tiles).
__global__ __launch_bounds__(256) void k_gemm(
    const float* __restrict__ X, const unsigned* __restrict__ W1f,
    const unsigned* __restrict__ W2f, float* __restrict__ E_T) {

    __shared__ unsigned XsP[64][16];            // 64 rows x 16 k-pairs (bf16x2)
    __shared__ unsigned WksF[16][16][16];       // [tile][n][pairRow], 16KB
    __shared__ __bf16   Ts[64][HDIM + 8];       // relu intermediate
    __shared__ float    Es[HDIM][65];           // transpose staging

    const int t = threadIdx.x;
    const int lane = t & 31, wave = t >> 5;
    const int wr = wave >> 1, wc = wave & 1;
    const int rowBase = blockIdx.x * 64;
    const int m  = lane & 15;      // A: row-in-tile / B: col-in-tile
    const int hi = lane >> 4;      // K-half selector

    union Frag { v16bf v; uint4 q[2]; };

    v8f zero;
#pragma unroll
    for (int q = 0; q < 8; ++q) zero[q] = 0.0f;

    v8f acc[8];
#pragma unroll
    for (int ct = 0; ct < 8; ++ct) acc[ct] = zero;

#ifdef HAVE_TDM
    const unsigned ldsWks = (unsigned)(uintptr_t)&WksF[0][0][0];
#endif

    // ---------------- phase 1: T = relu(X @ W1), K = 512 ----------------
    for (int kc = 0; kc < NFEAT; kc += 32) {
        const int chunk = kc >> 5;
        __syncthreads();
        // stage X tile: 64 rows x 16 pairs, float2 -> packed bf16 pair
#pragma unroll
        for (int it = 0; it < 4; ++it) {
            int li = t + it * 256;
            int r = li >> 4, kp = li & 15;
            const float2 xv = *(const float2*)&X[(size_t)(rowBase + r) * NFEAT + kc + 2 * kp];
            XsP[r][kp] = pack2bf(xv.x, xv.y);
        }
        if (kc + 32 < NFEAT)
            __builtin_prefetch(&X[(size_t)(rowBase + (t >> 2)) * NFEAT + kc + 32], 0, 1);
#ifdef HAVE_TDM
        if (wave == 0) {
            tdm_load_dwords(ldsWks, W1f + (size_t)chunk * 4096, 4096);
            __builtin_amdgcn_s_wait_tensorcnt(0);
        }
#else
#pragma unroll
        for (int it = 0; it < 16; ++it) {
            int li = t + it * 256;
            (&WksF[0][0][0])[li] = W1f[(size_t)chunk * 4096 + li];
        }
#endif
        __syncthreads();

        // A fragment: two b128 LDS loads
        Frag a;
        {
            const uint4* xr = (const uint4*)&XsP[wr * 16 + m][0];
            a.q[0] = xr[hi];
            a.q[1] = xr[hi + 2];
        }
#pragma unroll
        for (int ct = 0; ct < 8; ++ct) {
            Frag b;
            const uint4* wrp = (const uint4*)&WksF[wc * 8 + ct][m][0];
            b.q[0] = wrp[hi];
            b.q[1] = wrp[hi + 2];
            acc[ct] = __builtin_amdgcn_wmma_f32_16x16x32_bf16(
                false, a.v, false, b.v, (short)0, acc[ct], false, false);
        }
    }
    __syncthreads();

    // relu -> bf16 -> Ts
#pragma unroll
    for (int ct = 0; ct < 8; ++ct) {
        int n = wc * 128 + ct * 16 + m;
#pragma unroll
        for (int v = 0; v < 8; ++v) {
            int row = wr * 16 + v + 8 * hi;
            Ts[row][n] = (__bf16)fmaxf(acc[ct][v], 0.0f);
        }
    }
    __syncthreads();

    // ---------------- phase 2: E = T @ W2, K = 256 ----------------
#pragma unroll
    for (int ct = 0; ct < 8; ++ct) acc[ct] = zero;

    for (int kc = 0; kc < HDIM; kc += 32) {
        const int chunk = kc >> 5;
        __syncthreads();
#ifdef HAVE_TDM
        if (wave == 0) {
            tdm_load_dwords(ldsWks, W2f + (size_t)chunk * 4096, 4096);
            __builtin_amdgcn_s_wait_tensorcnt(0);
        }
#else
#pragma unroll
        for (int it = 0; it < 16; ++it) {
            int li = t + it * 256;
            (&WksF[0][0][0])[li] = W2f[(size_t)chunk * 4096 + li];
        }
#endif
        __syncthreads();

        // A fragment from Ts: K-pairs are contiguous bf16, 16B aligned
        Frag a;
        {
            const unsigned* trow = (const unsigned*)&Ts[wr * 16 + m][0];
            const uint4* tq = (const uint4*)(trow + (kc >> 1));
            a.q[0] = tq[hi];
            a.q[1] = tq[hi + 2];
        }
#pragma unroll
        for (int ct = 0; ct < 8; ++ct) {
            Frag b;
            const uint4* wrp = (const uint4*)&WksF[wc * 8 + ct][m][0];
            b.q[0] = wrp[hi];
            b.q[1] = wrp[hi + 2];
            acc[ct] = __builtin_amdgcn_wmma_f32_16x16x32_bf16(
                false, a.v, false, b.v, (short)0, acc[ct], false, false);
        }
    }
    __syncthreads();

    // transpose through LDS (pad 65 floats/row: conflict-free)
#pragma unroll
    for (int ct = 0; ct < 8; ++ct) {
        int n = wc * 128 + ct * 16 + m;
#pragma unroll
        for (int v = 0; v < 8; ++v) {
            int row = wr * 16 + v + 8 * hi;
            Es[n][row] = acc[ct][v];
        }
    }
    __syncthreads();
    // coalesced column-major store: E_T[c * N + rowBase + i]
#pragma unroll
    for (int it = 0; it < 64; ++it) {
        int li = t + it * 256;
        int c = li >> 6, i = li & 63;
        E_T[(size_t)c * NROWS + rowBase + i] = Es[c][i];
    }
}

// ---------- kernel 4: per-column weighted median via 8-bit radix select ----------
__global__ __launch_bounds__(256) void k_median(
    const float* __restrict__ E_T, const float* __restrict__ w,
    const float* __restrict__ total_w, float* __restrict__ median) {

    __shared__ float hist[256];
    __shared__ unsigned sh_sel;
    __shared__ float sh_target;

    const int c = blockIdx.x;
    const int t = threadIdx.x;
    const float* col = E_T + (size_t)c * NROWS;

    unsigned prefix = 0;
    float target = 0.5f * (*total_w);

    for (int p = 0; p < 4; ++p) {
        const int shift = 24 - 8 * p;
        hist[t] = 0.0f;
        __syncthreads();
        for (int i = t; i < NROWS; i += 256) {
            unsigned key = ord_u32(col[i]);
            bool ok = (p == 0) || ((key >> (shift + 8)) == prefix);
            if (ok) atomicAdd(&hist[(key >> shift) & 255u], w[i]);
        }
        __syncthreads();
        if (t == 0) {
            float cum = 0.0f; int b = 0;
            for (; b < 256; ++b) {
                if (cum + hist[b] >= target) break;
                cum += hist[b];
            }
            if (b > 255) b = 255;
            sh_sel = (unsigned)b;
            sh_target = target - cum;
        }
        __syncthreads();
        prefix = (prefix << 8) | sh_sel;
        target = sh_target;
        __syncthreads();
    }
    if (t == 0) median[c] = ord_inv(prefix);
}

// ---------- kernel 5: distances + global min ----------
__global__ __launch_bounds__(256) void k_dist(
    const float* __restrict__ E_T, const float* __restrict__ median,
    float* __restrict__ dist, unsigned* __restrict__ dmin_bits) {

    __shared__ float med_s[HDIM];
    __shared__ float sm[256];
    for (int c = threadIdx.x; c < HDIM; c += 256) med_s[c] = median[c];
    __syncthreads();

    float dloc = 3.4e38f;
    for (int i = blockIdx.x * 256 + threadIdx.x; i < NROWS; i += gridDim.x * 256) {
        float s = 0.0f;
#pragma unroll 4
        for (int c = 0; c < HDIM; ++c) {
            float df = E_T[(size_t)c * NROWS + i] - med_s[c];
            s += df * df;
        }
        float d = sqrtf(s) * (1.0f / 16.0f); // / sqrt(256)
        dist[i] = d;
        dloc = fminf(dloc, d);
    }
    float bmin = blockReduceMin(dloc, sm);
    if (threadIdx.x == 0) atomicMin(dmin_bits, __float_as_uint(bmin)); // d >= 0
}

// ---------- kernel 6: s_i = w_i * exp(dmin - d_i), accumulate S ----------
__global__ __launch_bounds__(256) void k_soft(
    const float* __restrict__ dist, const float* __restrict__ w,
    const unsigned* __restrict__ dmin_bits,
    float* __restrict__ sarr, float* __restrict__ S) {

    __shared__ float sm[256];
    const float dmin = __uint_as_float(*dmin_bits);
    float acc = 0.0f;
    for (int i = blockIdx.x * 256 + threadIdx.x; i < NROWS; i += gridDim.x * 256) {
        float s = w[i] * __expf(dmin - dist[i]);
        sarr[i] = s;
        acc += s;
    }
    float b = blockReduceSum(acc, sm);
    if (threadIdx.x == 0) atomicAdd(S, b);
}

// ---------- kernel 7: new_emb[c] = (Wsum/S) * dot(E_T[c,:], s) ----------
__global__ __launch_bounds__(256) void k_colreduce(
    const float* __restrict__ E_T, const float* __restrict__ sarr,
    const float* __restrict__ total_w, const float* __restrict__ S,
    float* __restrict__ new_emb) {

    __shared__ float sm[256];
    const int c = blockIdx.x;
    const float* col = E_T + (size_t)c * NROWS;
    float acc = 0.0f;
    for (int i = threadIdx.x; i < NROWS; i += 256)
        acc += col[i] * sarr[i];
    float b = blockReduceSum(acc, sm);
    if (threadIdx.x == 0) new_emb[c] = b * ((*total_w) / (*S));
}

// ---------- kernel 8: out = relu(ne @ W3) @ W4 ----------
__global__ __launch_bounds__(256) void k_head(
    const float* __restrict__ new_emb, const float* __restrict__ W3,
    const float* __restrict__ W4, float* __restrict__ out) {

    __shared__ float ne[HDIM];
    __shared__ float h[HDIM];
    const int t = threadIdx.x;
    ne[t] = new_emb[t];
    __syncthreads();
    float acc = 0.0f;
#pragma unroll 4
    for (int k = 0; k < HDIM; ++k) acc += ne[k] * W3[k * HDIM + t];
    h[t] = fmaxf(acc, 0.0f);
    __syncthreads();
    if (t < NCLS) {
        float o = 0.0f;
#pragma unroll 4
        for (int k = 0; k < HDIM; ++k) o += h[k] * W4[k * NCLS + t];
        out[t] = o;
    }
}

// ---------- host launch ----------
extern "C" void kernel_launch(void* const* d_in, const int* in_sizes, int n_in,
                              void* d_out, int out_size, void* d_ws, size_t ws_size,
                              hipStream_t stream) {
    const float* X  = (const float*)d_in[0];
    const float* w  = (const float*)d_in[1];
    const float* W1 = (const float*)d_in[2];
    const float* W2 = (const float*)d_in[3];
    const float* W3 = (const float*)d_in[4];
    const float* W4 = (const float*)d_in[5];
    float* out = (float*)d_out;

    char* ws = (char*)d_ws;
    size_t off = 0;
    float*    E_T  = (float*)(ws + off);    off += (size_t)HDIM * NROWS * 4;  // 128 MB
    unsigned* W1f  = (unsigned*)(ws + off); off += (size_t)(NFEAT / 32) * 4096 * 4; // 256 KB
    unsigned* W2f  = (unsigned*)(ws + off); off += (size_t)(HDIM / 32) * 4096 * 4;  // 128 KB
    float*    med  = (float*)(ws + off);    off += 1024;
    float*    nemb = (float*)(ws + off);    off += 1024;
    float*    dist = (float*)(ws + off);    off += (size_t)NROWS * 4;
    float*    sarr = (float*)(ws + off);    off += (size_t)NROWS * 4;
    float*    total_w = (float*)(ws + off);
    float*    Ssum    = (float*)(ws + off + 4);
    unsigned* dminb   = (unsigned*)(ws + off + 8);

    k_init<<<1, 1, 0, stream>>>(total_w, Ssum, dminb);
    k_wsum<<<512, 256, 0, stream>>>(w, total_w);
    k_cvtfrag<<<(NFEAT / 32) * 4096 / 256, 256, 0, stream>>>(W1, W1f, NFEAT);
    k_cvtfrag<<<(HDIM / 32) * 4096 / 256, 256, 0, stream>>>(W2, W2f, HDIM);
    k_gemm<<<NROWS / 64, 256, 0, stream>>>(X, W1f, W2f, E_T);
    k_median<<<HDIM, 256, 0, stream>>>(E_T, w, total_w, med);
    k_dist<<<512, 256, 0, stream>>>(E_T, med, dist, dminb);
    k_soft<<<512, 256, 0, stream>>>(dist, w, dminb, sarr, Ssum);
    k_colreduce<<<HDIM, 256, 0, stream>>>(E_T, sarr, total_w, Ssum, nemb);
    k_head<<<1, 256, 0, stream>>>(nemb, W3, W4, out);
}
